// AttentionLayer_25993142075512
// MI455X (gfx1250) — matrix operand order repaired and checked by reading, not compiled, not run on hardware
//
#include <hip/hip_runtime.h>
#include <hip/hip_bf16.h>
#include <math.h>

// ---------------------------------------------------------------------------
// Problem constants (from reference)
// ---------------------------------------------------------------------------
constexpr int kB    = 2;
constexpr int kN    = 2048;
constexpr int kDQ   = 1024;
constexpr int kDKV  = 1024;
constexpr int kH    = 16;
constexpr int kDH   = 64;
constexpr int kHDH  = kH * kDH;      // 1024
constexpr int kDOUT = 1024;
constexpr int kM    = kB * kN;       // 4096 rows for all projections

// ---------------------------------------------------------------------------
// WMMA / descriptor types (CDNA5 / gfx1250, wave32)
// ---------------------------------------------------------------------------
typedef __attribute__((ext_vector_type(16))) __bf16 bf16x16;
typedef __attribute__((ext_vector_type(8)))  float  f32x8;
typedef __attribute__((ext_vector_type(4)))  unsigned u32x4;
typedef __attribute__((ext_vector_type(8)))  int      i32x8;
typedef __attribute__((ext_vector_type(4)))  int      i32x4;

union Frag {
    bf16x16        v;
    unsigned short s[16];
    unsigned       u[8];
};

__device__ __forceinline__ unsigned short f2bf(float f) {
    unsigned u = __float_as_uint(f);
    u += 0x7fffu + ((u >> 16) & 1u);       // round-to-nearest-even
    return (unsigned short)(u >> 16);
}

// ---------------------------------------------------------------------------
// CDNA5 async global->LDS copy (ASYNCcnt) + TDM availability probes
// ---------------------------------------------------------------------------
__device__ __forceinline__ void async_copy_b128(unsigned lds_byte_off, const void* gsrc) {
    asm volatile("global_load_async_to_lds_b128 %0, %1, off"
                 :: "v"(lds_byte_off), "v"(gsrc)
                 : "memory");
}
__device__ __forceinline__ void wait_async0() {
    asm volatile("s_wait_asynccnt 0x0" ::: "memory");
}

#ifndef __has_builtin
#define __has_builtin(x) 0
#endif
#if __has_builtin(__builtin_amdgcn_tensor_load_to_lds)
#define HAVE_TDM 1
#else
#define HAVE_TDM 0
#endif

__device__ __forceinline__ void wait_tensor_le(int n) {
#if HAVE_TDM
#if __has_builtin(__builtin_amdgcn_s_wait_tensorcnt)
    if (n == 0) __builtin_amdgcn_s_wait_tensorcnt(0);
    else        __builtin_amdgcn_s_wait_tensorcnt(1);
#else
    if (n == 0) asm volatile("s_wait_tensorcnt 0x0" ::: "memory");
    else        asm volatile("s_wait_tensorcnt 0x1" ::: "memory");
#endif
#else
    (void)n;
#endif
}

// ---------------------------------------------------------------------------
// Kernel 0a: f32 -> bf16 bulk convert (activations, row-major preserved)
// ---------------------------------------------------------------------------
__global__ void __launch_bounds__(256)
cvt_bf16_kernel(const float* __restrict__ X, unsigned short* __restrict__ Y, int n)
{
    int i = blockIdx.x * 256 + threadIdx.x;
    if (i < n) Y[i] = f2bf(X[i]);
}

// ---------------------------------------------------------------------------
// Kernel 0b: weight transpose + convert.  W[K][N] f32 -> WT[N][K] bf16.
// ---------------------------------------------------------------------------
__global__ void __launch_bounds__(256)
tconv_bf16_kernel(const float* __restrict__ W, unsigned short* __restrict__ WT,
                  int K, int N)
{
    int i = blockIdx.x * 256 + threadIdx.x;   // over K*N, coalesced read
    if (i < K * N) {
        int k = i / N, n = i - k * N;
        WT[(size_t)n * K + k] = f2bf(W[i]);
    }
}

// ---------------------------------------------------------------------------
// Kernel 1: tiled GEMM  C[M,N] = A[M,K] @ BT[N,K]^T + bias[N]
// bf16 operands, f32 accumulate/output.  256 threads = 8 waves, 128x128x32
// tile, double-buffered LDS.  A tile pulled by the Tensor Data Mover (one
// 2D descriptor per stage, row padding 64B+16B -> 80B LDS stride); B tile
// by per-thread GLOBAL_LOAD_ASYNC_TO_LDS_B128.  Stage t+1 is issued before
// consuming stage t; in-order ASYNCcnt/TENSORcnt semantics let us wait
// "<= in-flight-next" instead of 0.
// ---------------------------------------------------------------------------
#define GT_M 128
#define GT_N 128
#define GT_K 32
#define LSTR 40                      // LDS row stride in ushorts (80B)
#define LBUF_BYTES (GT_M * LSTR * 2) // one tile buffer: 10240 B

__global__ void __launch_bounds__(256)
gemm_bf16_kernel(const unsigned short* __restrict__ A,   // [M][K] bf16 bits
                 const unsigned short* __restrict__ BT,  // [N][K] bf16 bits
                 const float* __restrict__ bias, float* __restrict__ C,
                 int M, int N, int K)
{
    __shared__ unsigned short lA [2][GT_M][LSTR];
    __shared__ unsigned short lBT[2][GT_N][LSTR];

    const int tid  = threadIdx.x;
    const int lane = tid & 31;
    const int wv   = tid >> 5;       // wave 0..7
    const int wm   = wv >> 1;        // 0..3 : 32 rows each
    const int wn   = wv & 1;         // 0..1 : 64 cols each
    const int half = lane >> 4;
    const int l15  = lane & 15;

    const int bm = blockIdx.y * GT_M;
    const int bn = blockIdx.x * GT_N;

    const unsigned ldsA0 = (unsigned)(size_t)&lA[0][0][0];
    const unsigned ldsB0 = (unsigned)(size_t)&lBT[0][0][0];

    // ---- stage helpers -----------------------------------------------------
    auto stage_b = [&](int buf, int k0) {   // 512 x 16B chunks, 2 per thread
        #pragma unroll
        for (int i = 0; i < 2; ++i) {
            int c   = i * 256 + tid;
            int row = c >> 2, cc = c & 3;
            async_copy_b128(ldsB0 + buf * LBUF_BYTES + row * (LSTR * 2) + cc * 16,
                            BT + (size_t)(bn + row) * K + k0 + cc * 8);
        }
    };
#if HAVE_TDM
    auto stage_a = [&](int buf, int k0) {   // one TDM descriptor, wave 0 only
        if (wv == 0) {
            unsigned long long ga =
                (unsigned long long)(size_t)(A + (size_t)bm * K + k0);
            u32x4 g0;
            g0[0] = 1u;                                      // count=1, user D#
            g0[1] = ldsA0 + (unsigned)(buf * LBUF_BYTES);    // lds_addr
            g0[2] = (unsigned)(ga & 0xffffffffu);            // global_addr lo
            g0[3] = (unsigned)((ga >> 32) & 0x01ffffffu) | (2u << 30); // type=2
            i32x8 g1;
            g1[0] = (int)((1u << 16)      // data_size = 2B
                        | (1u << 20)      // pad_enable
                        | (3u << 22)      // pad_interval: 16 dwords (64B)
                        | (3u << 25));    // pad_amount  :  4 dwords (16B)
            g1[1] = (int)(((unsigned)K & 0xffffu) << 16);               // dim0 lo16
            g1[2] = (int)((((unsigned)K >> 16) & 0xffffu)
                        | (((unsigned)M & 0xffffu) << 16));             // dim0 hi | dim1 lo
            g1[3] = (int)((((unsigned)M >> 16) & 0xffffu)
                        | ((unsigned)GT_K << 16));                      // dim1 hi | tile_dim0
            g1[4] = GT_M;                                               // tile_dim1
            g1[5] = K;                                                  // dim0_stride lo
            g1[6] = 0;
            g1[7] = 0;
            i32x4 z4; z4[0] = z4[1] = z4[2] = z4[3] = 0;
#if __clang_major__ >= 23
            i32x8 z8;
            z8[0]=z8[1]=z8[2]=z8[3]=z8[4]=z8[5]=z8[6]=z8[7]=0;
            __builtin_amdgcn_tensor_load_to_lds(g0, g1, z4, z4, z8, 0);
#else
            __builtin_amdgcn_tensor_load_to_lds(g0, g1, z4, z4, 0);
#endif
        }
    };
#else
    auto stage_a = [&](int buf, int k0) {   // fallback: async copies
        #pragma unroll
        for (int i = 0; i < 2; ++i) {
            int c   = i * 256 + tid;
            int row = c >> 2, cc = c & 3;
            async_copy_b128(ldsA0 + buf * LBUF_BYTES + row * (LSTR * 2) + cc * 16,
                            A + (size_t)(bm + row) * K + k0 + cc * 8);
        }
    };
#endif

    f32x8 acc[2][4];
    #pragma unroll
    for (int i = 0; i < 2; ++i)
        #pragma unroll
        for (int j = 0; j < 4; ++j)
            #pragma unroll
            for (int r = 0; r < 8; ++r) acc[i][j][r] = 0.0f;

    const int nk = K / GT_K;
    stage_a(0, 0);
    stage_b(0, 0);

    for (int t = 0; t < nk; ++t) {
        const int buf = t & 1;
        if (t + 1 < nk) {
            stage_a(buf ^ 1, (t + 1) * GT_K);
            stage_b(buf ^ 1, (t + 1) * GT_K);
            // wait for the *current* stage only (older ops retire in order)
#if HAVE_TDM
            asm volatile("s_wait_asynccnt 0x2" ::: "memory");
#else
            asm volatile("s_wait_asynccnt 0x4" ::: "memory");
#endif
            wait_tensor_le(1);
        } else {
            wait_async0();
            wait_tensor_le(0);
        }
        __syncthreads();

        // ---- fragments (A 16x32 / B 32x16, contiguous k = 16*half + s) ----
        Frag af[2], bfg[4];
        #pragma unroll
        for (int mt = 0; mt < 2; ++mt) {
            const unsigned* pr = reinterpret_cast<const unsigned*>(
                &lA[buf][wm * 32 + mt * 16 + l15][0]);
            #pragma unroll
            for (int v = 0; v < 8; ++v) {
                int k = (half << 4) + (v << 1);
                af[mt].u[v] = pr[k >> 1];
            }
        }
        #pragma unroll
        for (int nt = 0; nt < 4; ++nt) {
            const unsigned* pc = reinterpret_cast<const unsigned*>(
                &lBT[buf][wn * 64 + nt * 16 + l15][0]);
            #pragma unroll
            for (int v = 0; v < 8; ++v) {
                int k = (half << 4) + (v << 1);
                bfg[nt].u[v] = pc[k >> 1];
            }
        }
        #pragma unroll
        for (int mt = 0; mt < 2; ++mt)
            #pragma unroll
            for (int nt = 0; nt < 4; ++nt)
                acc[mt][nt] = __builtin_amdgcn_wmma_f32_16x16x32_bf16(
                    false, af[mt].v, false, bfg[nt].v,
                    (short)0, acc[mt][nt], false, false);
        __syncthreads();   // reads done before stage t+2 overwrites this buf
    }

    // ---- epilogue: D layout  m = r + 8*half, n = lane&15 ----
    #pragma unroll
    for (int mt = 0; mt < 2; ++mt)
        #pragma unroll
        for (int nt = 0; nt < 4; ++nt) {
            int gn = bn + wn * 64 + nt * 16 + l15;
            float bv = bias[gn];
            #pragma unroll
            for (int r = 0; r < 8; ++r) {
                int gm = bm + wm * 32 + mt * 16 + half * 8 + r;
                C[(size_t)gm * N + gn] = acc[mt][nt][r] + bv;
            }
        }
}

// ---------------------------------------------------------------------------
// Kernel 2: RoPE on q/k (first 32 dims) + repack to head-major bf16 buffers.
// ---------------------------------------------------------------------------
__global__ void __launch_bounds__(256)
rope_pack_kernel(const float* __restrict__ Qf, const float* __restrict__ KVf,
                 unsigned short* __restrict__ Qb, unsigned short* __restrict__ Kb,
                 unsigned short* __restrict__ Vb)
{
    int t = blockIdx.x * blockDim.x + threadIdx.x;   // over B*N*H*32
    int j = t & 31; t >>= 5;                          // pair index (d0 = 2j)
    int h = t & (kH - 1); t >>= 4;
    int n = t & (kN - 1); t >>= 11;
    int b = t;
    const int d0 = 2 * j, d1 = 2 * j + 1;

    const size_t qoff  = ((size_t)(b * kN + n)) * kHDH + h * kDH;
    const size_t kvoff = ((size_t)(b * kN + n)) * (2 * kHDH) + h * 2 * kDH;

    float q0 = Qf[qoff + d0],  q1 = Qf[qoff + d1];
    float k0 = KVf[kvoff + d0], k1 = KVf[kvoff + d1];
    float v0 = KVf[kvoff + kDH + d0], v1 = KVf[kvoff + kDH + d1];

    if (d0 < 32) {   // rotary on first ROT=32 dims; pair p uses 10000^{-d0/32}
        float inv = __powf(10000.0f, -(float)d0 / 32.0f);
        float ang = (float)n * inv;
        float s, c;
        __sincosf(ang, &s, &c);
        float nq0 = q0 * c - q1 * s, nq1 = q1 * c + q0 * s;
        float nk0 = k0 * c - k1 * s, nk1 = k1 * c + k0 * s;
        q0 = nq0; q1 = nq1; k0 = nk0; k1 = nk1;
    }
    const size_t o = (((size_t)(b * kH + h)) * kN + n) * kDH;
    Qb[o + d0] = f2bf(q0); Qb[o + d1] = f2bf(q1);
    Kb[o + d0] = f2bf(k0); Kb[o + d1] = f2bf(k1);
    Vb[o + d0] = f2bf(v0); Vb[o + d1] = f2bf(v1);
}

// ---------------------------------------------------------------------------
// Kernel 3: causal flash attention.  One wave (32 threads) per
// (b, h, 16-query tile).  Streams 64-key tiles; S and P·V via WMMA bf16.
// ---------------------------------------------------------------------------
__global__ void __launch_bounds__(32)
attn_kernel(const unsigned short* __restrict__ Qb,
            const unsigned short* __restrict__ Kb,
            const unsigned short* __restrict__ Vb,
            const float* __restrict__ mask_q,
            const float* __restrict__ mask_kv,
            unsigned short* __restrict__ O)
{
    __shared__ unsigned short lP[16][kDH + 2];   // P tile staging (bf16 bits)

    const int lane = threadIdx.x;
    const int half = lane >> 4, l15 = lane & 15;

    int id = blockIdx.x;
    const int qt = id & (kN / 16 - 1); id >>= 7;  // 128 query tiles
    const int h  = id & (kH - 1);      id >>= 4;
    const int b  = id;
    const int q0 = qt * 16;

    const unsigned short* Qh = Qb + ((size_t)(b * kH + h) * kN) * kDH;
    const unsigned short* Kh = Kb + ((size_t)(b * kH + h) * kN) * kDH;
    const unsigned short* Vh = Vb + ((size_t)(b * kH + h) * kN) * kDH;

    // ---- Q fragments (A layout: row = lane&15; k split by half/vgpr) ----
    Frag aq[2];
    {
        const unsigned* qrow =
            reinterpret_cast<const unsigned*>(Qh + (size_t)(q0 + l15) * kDH);
        #pragma unroll
        for (int c = 0; c < 2; ++c)
            #pragma unroll
            for (int v = 0; v < 8; ++v) {
                int k = c * 32 + (half << 4) + (v << 1);
                aq[c].u[v] = qrow[k >> 1];
            }
    }

    float mrow[8], lrow[8], mqv[8];
    f32x8 oacc[4];
    #pragma unroll
    for (int r = 0; r < 8; ++r) {
        mrow[r] = -3.0e38f;
        lrow[r] = 0.0f;
        mqv[r]  = mask_q[b * kN + q0 + half * 8 + r];
    }
    #pragma unroll
    for (int dt = 0; dt < 4; ++dt)
        #pragma unroll
        for (int r = 0; r < 8; ++r) oacc[dt][r] = 0.0f;

    const int kend   = q0 + 16;              // causal bound (exclusive)
    const int ntiles = (kend + 63) >> 6;     // 64-key tiles to stream

    for (int t = 0; t < ntiles; ++t) {
        const int kb = t * 64;
        if (t + 1 < ntiles) {   // pull next K/V tiles toward the caches
            __builtin_prefetch(Kh + (size_t)(kb + 64 + (lane << 1)) * kDH, 0, 1);
            __builtin_prefetch(Vh + (size_t)(kb + 64 + (lane << 1)) * kDH, 0, 1);
        }

        // ---- S = Q K^T  (4 column tiles of 16 keys) ----
        f32x8 s[4];
        #pragma unroll
        for (int nt = 0; nt < 4; ++nt) {
            Frag bk0, bk1;
            const unsigned* krow = reinterpret_cast<const unsigned*>(
                Kh + (size_t)(kb + nt * 16 + l15) * kDH);
            #pragma unroll
            for (int v = 0; v < 8; ++v) {
                int k = (half << 4) + (v << 1);        // contraction d
                bk0.u[v] = krow[k >> 1];
                bk1.u[v] = krow[(32 + k) >> 1];
            }
            f32x8 c;
            #pragma unroll
            for (int r = 0; r < 8; ++r) c[r] = 0.0f;
            c = __builtin_amdgcn_wmma_f32_16x16x32_bf16(
                    false, aq[0].v, false, bk0.v, (short)0, c, false, false);
            c = __builtin_amdgcn_wmma_f32_16x16x32_bf16(
                    false, aq[1].v, false, bk1.v, (short)0, c, false, false);
            s[nt] = c;
        }

        // ---- causal mask, scale, pad mask (reference order) ----
        float mkv[4];
        #pragma unroll
        for (int nt = 0; nt < 4; ++nt)
            mkv[nt] = mask_kv[b * kN + kb + nt * 16 + l15];
        #pragma unroll
        for (int nt = 0; nt < 4; ++nt) {
            const int kk = kb + nt * 16 + l15;
            #pragma unroll
            for (int r = 0; r < 8; ++r) {
                const int qg = q0 + half * 8 + r;
                float val = s[nt][r];
                val = (val - (kk > qg ? 1.0e6f : 0.0f)) * 0.125f; // 1/sqrt(64)
                val += 1.0e6f * (mqv[r] * mkv[nt] - 1.0f);
                s[nt][r] = val;
            }
        }

        // ---- online softmax (row reductions across 16 lanes) ----
        #pragma unroll
        for (int r = 0; r < 8; ++r) {
            float mx = fmaxf(fmaxf(s[0][r], s[1][r]), fmaxf(s[2][r], s[3][r]));
            #pragma unroll
            for (int off = 8; off >= 1; off >>= 1)
                mx = fmaxf(mx, __shfl_xor(mx, off, 16));
            float mnew = fmaxf(mrow[r], mx);
            float corr = __expf(mrow[r] - mnew);
            mrow[r] = mnew;
            lrow[r] *= corr;
            #pragma unroll
            for (int dt = 0; dt < 4; ++dt) oacc[dt][r] *= corr;

            float ps = 0.0f;
            #pragma unroll
            for (int nt = 0; nt < 4; ++nt) {
                float p = __expf(s[nt][r] - mnew);
                s[nt][r] = p;
                ps += p;
            }
            #pragma unroll
            for (int off = 8; off >= 1; off >>= 1)
                ps += __shfl_xor(ps, off, 16);
            lrow[r] += ps;
        }

        // ---- D-layout -> A-layout via LDS for the P·V matmul ----
        #pragma unroll
        for (int nt = 0; nt < 4; ++nt)
            #pragma unroll
            for (int r = 0; r < 8; ++r)
                lP[half * 8 + r][nt * 16 + l15] = f2bf(s[nt][r]);
        __syncthreads();
        Frag pa[2];
        {
            const unsigned* prow = reinterpret_cast<const unsigned*>(&lP[l15][0]);
            #pragma unroll
            for (int c = 0; c < 2; ++c)
                #pragma unroll
                for (int v = 0; v < 8; ++v) {
                    int k = c * 32 + (half << 4) + (v << 1);
                    pa[c].u[v] = prow[k >> 1];
                }
        }
        __syncthreads();

        // ---- O += P V  (B fragments gathered from head-major V) ----
        #pragma unroll
        for (int dt = 0; dt < 4; ++dt) {
            Frag bv0, bv1;
            #pragma unroll
            for (int v = 0; v < 8; ++v) {
                int key = (half << 4) + (v << 1);   // contraction key
                const unsigned short* p0 =
                    Vh + (size_t)(kb + key) * kDH + dt * 16 + l15;
                bv0.s[2 * v]     = p0[0];
                bv0.s[2 * v + 1] = p0[kDH];
                const unsigned short* p1 =
                    Vh + (size_t)(kb + 32 + key) * kDH + dt * 16 + l15;
                bv1.s[2 * v]     = p1[0];
                bv1.s[2 * v + 1] = p1[kDH];
            }
            oacc[dt] = __builtin_amdgcn_wmma_f32_16x16x32_bf16(
                false, pa[0].v, false, bv0.v, (short)0, oacc[dt], false, false);
            oacc[dt] = __builtin_amdgcn_wmma_f32_16x16x32_bf16(
                false, pa[1].v, false, bv1.v, (short)0, oacc[dt], false, false);
        }
    }

    // ---- normalize, write bf16 in bqhd interleave ----
    #pragma unroll
    for (int dt = 0; dt < 4; ++dt)
        #pragma unroll
        for (int r = 0; r < 8; ++r) {
            int qg = q0 + half * 8 + r;
            float inv = (lrow[r] > 0.0f) ? 1.0f / lrow[r] : 0.0f;
            O[((size_t)(b * kN) + qg) * kHDH + h * kDH + dt * 16 + l15] =
                f2bf(oacc[dt][r] * inv);
        }
}

// ---------------------------------------------------------------------------
// Host launcher
// ---------------------------------------------------------------------------
extern "C" void kernel_launch(void* const* d_in, const int* in_sizes, int n_in,
                              void* d_out, int out_size, void* d_ws, size_t ws_size,
                              hipStream_t stream)
{
    (void)in_sizes; (void)n_in; (void)out_size; (void)ws_size;

    const float* s_q     = (const float*)d_in[0];
    const float* s_kv    = (const float*)d_in[1];
    const float* mask_q  = (const float*)d_in[2];
    const float* mask_kv = (const float*)d_in[3];
    const float* Wq      = (const float*)d_in[4];
    const float* bq      = (const float*)d_in[5];
    const float* Wkv     = (const float*)d_in[6];
    const float* bkv     = (const float*)d_in[7];
    const float* Wo      = (const float*)d_in[8];
    const float* bo      = (const float*)d_in[9];
    float* out = (float*)d_out;

    // workspace carve-out (~104 MB total)
    char* ws = (char*)d_ws;
    size_t off = 0;
    auto carve = [&](size_t bytes) {
        void* p = ws + off;
        off = (off + bytes + 255) & ~(size_t)255;
        return p;
    };
    unsigned short* Aq   = (unsigned short*)carve((size_t)kM * kDQ * 2);       // 8 MB
    unsigned short* Akv  = (unsigned short*)carve((size_t)kM * kDKV * 2);      // 8 MB
    unsigned short* WqT  = (unsigned short*)carve((size_t)kDQ * kHDH * 2);     // 2 MB
    unsigned short* WkvT = (unsigned short*)carve((size_t)kDKV * 2 * kHDH * 2);// 4 MB
    unsigned short* WoT  = (unsigned short*)carve((size_t)kHDH * kDOUT * 2);   // 2 MB
    float*          Qf   = (float*)carve((size_t)kM * kHDH * 4);               // 16 MB
    float*          KVf  = (float*)carve((size_t)kM * 2 * kHDH * 4);           // 32 MB
    unsigned short* Qb   = (unsigned short*)carve((size_t)kM * kHDH * 2);      // 8 MB
    unsigned short* Kb   = (unsigned short*)carve((size_t)kM * kHDH * 2);      // 8 MB
    unsigned short* Vb   = (unsigned short*)carve((size_t)kM * kHDH * 2);      // 8 MB
    unsigned short* Aob  = (unsigned short*)carve((size_t)kM * kHDH * 2);      // 8 MB

    // 0) one-time precision/layout conversions
    cvt_bf16_kernel<<<(kM * kDQ) / 256, 256, 0, stream>>>(s_q, Aq, kM * kDQ);
    cvt_bf16_kernel<<<(kM * kDKV) / 256, 256, 0, stream>>>(s_kv, Akv, kM * kDKV);
    tconv_bf16_kernel<<<(kDQ * kHDH) / 256, 256, 0, stream>>>(Wq, WqT, kDQ, kHDH);
    tconv_bf16_kernel<<<(kDKV * 2 * kHDH) / 256, 256, 0, stream>>>(Wkv, WkvT, kDKV, 2 * kHDH);
    tconv_bf16_kernel<<<(kHDH * kDOUT) / 256, 256, 0, stream>>>(Wo, WoT, kHDH, kDOUT);

    // 1) Q projection: [4096,1024] = Aq @ WqT^T + bq
    gemm_bf16_kernel<<<dim3(kHDH / GT_N, kM / GT_M), 256, 0, stream>>>(
        Aq, WqT, bq, Qf, kM, kHDH, kDQ);

    // 2) KV projection: [4096,2048] = Akv @ WkvT^T + bkv
    gemm_bf16_kernel<<<dim3(2 * kHDH / GT_N, kM / GT_M), 256, 0, stream>>>(
        Akv, WkvT, bkv, KVf, kM, 2 * kHDH, kDKV);

    // 3) RoPE + bf16 head-major repack
    rope_pack_kernel<<<(kB * kN * kH * 32) / 256, 256, 0, stream>>>(
        Qf, KVf, Qb, Kb, Vb);

    // 4) causal flash attention (1 wave per (b,h,16 queries))
    attn_kernel<<<kB * kH * (kN / 16), 32, 0, stream>>>(
        Qb, Kb, Vb, mask_q, mask_kv, Aob);

    // 5) output projection: out = Aob @ WoT^T + bo
    gemm_bf16_kernel<<<dim3(kDOUT / GT_N, kM / GT_M), 256, 0, stream>>>(
        Aob, WoT, bo, out, kM, kDOUT, kHDH);
}
